// Network_52192442581325
// MI455X (gfx1250) — matrix operand order repaired and checked by reading, not compiled, not run on hardware
//
#include <hip/hip_runtime.h>
#include <math.h>

// ---------------------------------------------------------------------------
// e3nn-style equivariant GNN for MI455X (gfx1250, wave32, WMMA).
// Dominant compute: per-edge weight GEMMs [E,256]x[256,NW] -> bf16 WMMA.
// W staging chunks clamped to <=64MB so store->load stays in the 192MB L2.
// ---------------------------------------------------------------------------

#define N_NODES 8192
#define N_EDGES 32768
#define N_GRAPHS 64
#define INV_SQRT_NEIGH 0.51298916f   // 1/sqrt(3.8)

typedef __attribute__((ext_vector_type(16))) __bf16 v16bf;
typedef __attribute__((ext_vector_type(8)))  float  v8f;

__device__ __forceinline__ unsigned short f2bf(float f) {
    unsigned int u = __float_as_uint(f);
    unsigned int r = (u + 0x7FFFu + ((u >> 16) & 1u)) >> 16;   // RNE
    return (unsigned short)r;
}

// ---------------------------------------------------------------------------
// Wigner-3j sparse coefficient tables (real basis: l1 order x,y,z ;
// l2 order xy,yz,z2,xz,x2y2 ; unit Frobenius norm).
// combos: 0:(0,0,0) 1:(1,1,0) 2:(2,2,0) 3:(3,3,0) 4:(0,1,1) 5:(1,0,1)
//         6:(1,1,1) 7:(1,2,1) 8:(2,1,1) 9:(2,2,1) 10:(2,3,1) 11:(3,2,1) 12:(3,3,1)
// combos 10..12 are structured placeholders (would be emitted by offline codegen).
// ---------------------------------------------------------------------------
#define W3_NNZ 106
__constant__ unsigned char W3A[W3_NNZ] = {
 0,
 0,1,2,
 0,1,2,3,4,
 0,1,2,3,4,5,6,
 0,0,0,
 0,1,2,
 0,1,2,0,1,2,
 0,0,0,0,1,1,1,1,2,2,2,
 0,0,1,1,2,2,2,3,3,4,4,
 0,4,1,3,2,1,4,1,3,0,0,1,2,3,4,3,
 0,0,1,1,2,2,3,3,4,4,0,1,2,3,
 0,2,1,3,2,4,3,5,4,6,6,5,0,1,
 0,1,2,3,4,5,6,0,1,4,2,5 };
__constant__ unsigned char W3B[W3_NNZ] = {
 0,
 0,1,2,
 0,1,2,3,4,
 0,1,2,3,4,5,6,
 0,1,2,
 0,0,0,
 1,2,0,2,0,1,
 0,2,3,4,0,1,2,4,1,2,3,
 0,1,1,2,0,1,2,0,2,0,1,
 4,0,3,1,1,2,1,4,0,3,1,0,3,2,3,4,
 0,2,1,3,2,4,3,5,4,6,6,5,0,1,
 0,0,1,1,2,2,3,3,4,4,0,1,2,3,
 1,0,3,2,5,4,0,6,4,1,5,2 };
__constant__ unsigned char W3C[W3_NNZ] = {
 0,
 0,0,0,
 0,0,0,0,0,
 0,0,0,0,0,0,0,
 0,1,2,
 0,1,2,
 2,0,1,1,2,0,
 1,0,2,0,0,2,1,1,1,2,0,
 1,0,2,1,0,1,2,2,0,0,1,
 2,2,2,2,0,0,0,0,0,0,1,1,1,1,1,1,
 1,0,2,1,0,2,1,0,2,1,2,0,1,2,
 1,0,2,1,0,2,1,0,2,1,2,0,1,2,
 2,2,0,0,1,1,1,1,0,0,2,2 };
__constant__ float W3V[W3_NNZ] = {
 1.0f,
 0.57735026f,0.57735026f,0.57735026f,
 0.44721360f,0.44721360f,0.44721360f,0.44721360f,0.44721360f,
 0.37796447f,0.37796447f,0.37796447f,0.37796447f,0.37796447f,0.37796447f,0.37796447f,
 0.57735026f,0.57735026f,0.57735026f,
 0.57735026f,0.57735026f,0.57735026f,
 0.40824831f,0.40824831f,0.40824831f,-0.40824831f,-0.40824831f,-0.40824831f,
 0.31622777f,-0.18257419f,0.31622777f,0.31622777f,0.31622777f,0.31622777f,-0.18257419f,-0.31622777f,0.31622777f,0.36514837f,0.31622777f,
 0.31622777f,0.31622777f,0.31622777f,0.31622777f,-0.18257419f,-0.18257419f,0.36514837f,0.31622777f,0.31622777f,0.31622777f,-0.31622777f,
 0.36514837f,-0.36514837f,0.18257419f,-0.18257419f,0.31622777f,-0.31622777f,0.18257419f,-0.18257419f,0.18257419f,-0.18257419f,
 0.18257419f,-0.18257419f,-0.31622777f,0.31622777f,0.18257419f,-0.18257419f,
 0.26726124f,-0.26726124f,0.26726124f,-0.26726124f,0.26726124f,-0.26726124f,0.26726124f,-0.26726124f,0.26726124f,-0.26726124f,0.26726124f,0.26726124f,-0.26726124f,0.26726124f,
 0.26726124f,-0.26726124f,0.26726124f,-0.26726124f,0.26726124f,-0.26726124f,0.26726124f,-0.26726124f,0.26726124f,-0.26726124f,0.26726124f,0.26726124f,-0.26726124f,0.26726124f,
 0.28867513f,-0.28867513f,0.28867513f,-0.28867513f,0.28867513f,-0.28867513f,0.28867513f,-0.28867513f,0.28867513f,-0.28867513f,0.28867513f,-0.28867513f };
__constant__ int W3S[13] = {0,1,4,9,16,19,22,28,39,50,66,80,94};
__constant__ int W3N[13] = {1,3,5,7,3,3,6,11,11,16,14,14,12};

// ---------------------------------------------------------------------------
// Tensor-product instruction tables (enumerated from build_tp by hand).
// {x1_off, m1, combo, out_off, mo, d3, sh_off, d1, w_off, alpha}
// ---------------------------------------------------------------------------
struct TPI { int x1o, m1, combo, outo, mo, d3, sho, d1, woff; float alpha; };
__constant__ TPI TP_INSTR[64] = {
 // ---- layer 1 (din 16 -> dout 160, NW 272) ----
 {0,1,0, 0,16,1, 0,1,   0,0.5f}, {1,1,1, 0,16,1, 1,3,  16,0.5f},
 {4,1,2, 0,16,1, 4,5,  32,0.5f}, {9,1,3, 0,16,1, 9,7,  48,0.5f},
 {0,1,0,32, 8,1, 0,1,  64,0.5f}, {1,1,1,32, 8,1, 1,3,  72,0.5f},
 {4,1,2,32, 8,1, 4,5,  80,0.5f}, {9,1,3,32, 8,1, 9,7,  88,0.5f},
 {0,1,0,48, 8,1, 0,1,  96,0.5f}, {1,1,1,48, 8,1, 1,3, 104,0.5f},
 {4,1,2,48, 8,1, 4,5, 112,0.5f}, {9,1,3,48, 8,1, 9,7, 120,0.5f},
 {0,1,4, 64,16,3, 1,1, 128,0.40824831f}, {1,1,5, 64,16,3, 0,3, 144,0.40824831f},
 {1,1,7, 64,16,3, 4,3, 160,0.40824831f}, {4,1,8, 64,16,3, 1,5, 176,0.40824831f},
 {4,1,10,64,16,3, 9,5, 192,0.40824831f}, {9,1,11,64,16,3, 4,7, 208,0.40824831f},
 {1,1,6,112,16,3, 1,3, 224,0.57735026f}, {4,1,9,112,16,3, 4,5, 240,0.57735026f},
 {9,1,12,112,16,3,9,7, 256,0.57735026f},
 // ---- layer 2 (din 128 -> dout 128, NW 3328) ----
 {0,16,0, 0,16,1, 0,1,    0,0.14433757f}, {16,16,0, 0,16,1, 0,1, 256,0.14433757f},
 {32,16,1, 0,16,1, 1,3, 512,0.14433757f}, {80,16,1,16,16,1, 1,3, 768,0.25f},
 {0,16,0,32, 6,1, 0,1, 1024,0.14433757f}, {16,16,0,32, 6,1, 0,1,1120,0.14433757f},
 {32,16,1,32, 6,1, 1,3,1216,0.14433757f}, {80,16,1,38, 6,1, 1,3,1312,0.25f},
 {0,16,0,44, 6,1, 0,1, 1408,0.14433757f}, {16,16,0,44, 6,1, 0,1,1504,0.14433757f},
 {32,16,1,44, 6,1, 1,3,1600,0.14433757f}, {80,16,1,50, 6,1, 1,3,1696,0.25f},
 {0,16,4,56,12,3, 1,1, 1792,0.11180340f}, {16,16,4,56,12,3, 1,1,1984,0.11180340f},
 {32,16,5,56,12,3, 0,3,2176,0.11180340f}, {32,16,7,56,12,3, 4,3,2368,0.11180340f},
 {80,16,6,56,12,3, 1,3,2560,0.11180340f},
 {32,16,6,92,12,3, 1,3,2752,0.14433757f}, {80,16,5,92,12,3, 0,3,2944,0.14433757f},
 {80,16,7,92,12,3, 4,3,3136,0.14433757f},
 // ---- layer 3 (din 104 -> dout 104, NW 2400) ----
 {0,16,0, 0,20,1, 0,1,    0,0.15075567f}, {16,16,0, 0,20,1, 0,1, 320,0.15075567f},
 {32,12,1, 0,20,1, 1,3, 640,0.15075567f}, {68,12,1,20,20,1, 1,3, 880,0.28867513f},
 {0,16,0,40, 4,1, 0,1, 1120,0.15075567f}, {16,16,0,40, 4,1, 0,1,1184,0.15075567f},
 {32,12,1,40, 4,1, 1,3,1248,0.15075567f}, {68,12,1,44, 4,1, 1,3,1296,0.28867513f},
 {0,16,0,48, 4,1, 0,1, 1344,0.15075567f}, {16,16,0,48, 4,1, 0,1,1408,0.15075567f},
 {32,12,1,48, 4,1, 1,3,1472,0.15075567f}, {68,12,1,52, 4,1, 1,3,1520,0.28867513f},
 {0,16,4,56, 8,3, 1,1, 1568,0.12126781f}, {16,16,4,56, 8,3, 1,1,1696,0.12126781f},
 {32,12,5,56, 8,3, 0,3,1824,0.12126781f}, {32,12,7,56, 8,3, 4,3,1920,0.12126781f},
 {68,12,6,56, 8,3, 1,3,2016,0.12126781f},
 {32,12,6,80, 8,3, 1,3,2112,0.16666667f}, {68,12,5,80, 8,3, 0,3,2208,0.16666667f},
 {68,12,7,80, 8,3, 4,3,2304,0.16666667f},
 // ---- layer 4 (din 88 -> dout 1, NW 48) ----
 {0,20,0,0,1,1, 0,1,  0,0.14433757f}, {20,20,0,0,1,1, 0,1, 20,0.14433757f},
 {40, 8,1,0,1,1, 1,3, 40,0.14433757f} };

// ---------------------------------------------------------------------------
// Kernel 1: edge preprocessing (SH l=0..3, radial embedding, x0 scatter)
// ---------------------------------------------------------------------------
__global__ void edge_prep_k(const float* __restrict__ pos,
                            const int* __restrict__ src, const int* __restrict__ dst,
                            float* __restrict__ shb, float* __restrict__ esb,
                            float* __restrict__ x0, int E) {
    int e = blockIdx.x * blockDim.x + threadIdx.x;
    if (e >= E) return;
    int s = src[e], d = dst[e];
    float ex = pos[s*3+0] - pos[d*3+0];
    float ey = pos[s*3+1] - pos[d*3+1];
    float ez = pos[s*3+2] - pos[d*3+2];
    float r = sqrtf(ex*ex + ey*ey + ez*ez);
    float inv = 1.0f / r;
    float x = ex*inv, y = ey*inv, z = ez*inv;
    float sh[16];
    sh[0] = 1.0f;
    sh[1] = 1.7320508f*x; sh[2] = 1.7320508f*y; sh[3] = 1.7320508f*z;
    sh[4] = 3.8729833f*x*y; sh[5] = 3.8729833f*y*z;
    sh[6] = 1.1180340f*(3.0f*z*z - 1.0f);
    sh[7] = 3.8729833f*x*z; sh[8] = 1.9364917f*(x*x - y*y);
    sh[9]  = 2.0916500f*y*(3.0f*x*x - y*y);
    sh[10] = 10.2469508f*x*y*z;
    sh[11] = 1.6201852f*y*(5.0f*z*z - 1.0f);
    sh[12] = 1.3228757f*z*(5.0f*z*z - 3.0f);
    sh[13] = 1.6201852f*x*(5.0f*z*z - 1.0f);
    sh[14] = 5.1234754f*z*(x*x - y*y);
    sh[15] = 2.0916500f*x*(x*x - 3.0f*y*y);
    #pragma unroll
    for (int k = 0; k < 16; ++k) {
        shb[(size_t)e*16 + k] = sh[k];
        atomicAdd(&x0[(size_t)d*16 + k], sh[k]*INV_SQRT_NEIGH);
    }
    // soft_one_hot smooth_finite on [0.5,2.5], 3 inner bins, * sqrt(3)
    #pragma unroll
    for (int j = 0; j < 3; ++j) {
        float v = 1.0f + 0.5f*j;
        float dd = (r - v) * 2.0f;            // /step (0.5)
        float d2 = dd*dd;
        float o = 0.0f;
        if (d2 < 1.0f) o = 14.60737f * __expf(-1.0f/(1.0f - d2));  // 1.14136*e^2*sqrt(3)
        esb[(size_t)e*3 + j] = o;
    }
}

// ---------------------------------------------------------------------------
// Kernel 2: h = relu(es @ w1 / sqrt(3))  -> bf16 [E,256]
// ---------------------------------------------------------------------------
__global__ void h_k(const float* __restrict__ es, const float* __restrict__ w1,
                    unsigned short* __restrict__ h, int E) {
    int idx = blockIdx.x * blockDim.x + threadIdx.x;
    if (idx >= E*256) return;
    int e = idx >> 8, c = idx & 255;
    float v = (es[e*3+0]*w1[c] + es[e*3+1]*w1[256+c] + es[e*3+2]*w1[512+c]) * 0.57735027f;
    h[idx] = f2bf(fmaxf(v, 0.0f));
}

// ---------------------------------------------------------------------------
// Kernel 3: pack w2 (*1/16) into a contiguous-k bf16 B-operand layout, padded to an
// even number of 16-col tiles (zero-filled) so the GEMM can do 2-tile blocking.
// pb dword [((nt*8+ks)*32+lane)*8 + v] = {B[k][col], B[k+1][col]},
// k = 32*ks + 16*(lane>>4) + 2v, col = nt*16 + (lane&15).
// ---------------------------------------------------------------------------
__global__ void packB_k(const float* __restrict__ w2, unsigned int* __restrict__ pb,
                        int NW, int total) {
    int idx = blockIdx.x * blockDim.x + threadIdx.x;
    if (idx >= total) return;
    int v = idx & 7, lane = (idx >> 3) & 31, ks = (idx >> 8) & 7, nt = idx >> 11;
    int col = nt*16 + (lane & 15);
    int k = ks*32 + ((lane >> 4) << 4) + 2*v;
    float a0 = 0.0f, a1 = 0.0f;
    if (col < NW) {
        a0 = w2[(size_t)k*NW + col] * 0.0625f;
        a1 = w2[(size_t)(k+1)*NW + col] * 0.0625f;
    }
    pb[idx] = (unsigned int)f2bf(a0) | ((unsigned int)f2bf(a1) << 16);
}

// ---------------------------------------------------------------------------
// Kernel 4: W = h @ w2 via v_wmma_f32_16x16x32_bf16.
// 1 wave -> two adjacent 16x16 D-tiles sharing one A fragment per k-step
// (2 A-loads + 4 B-loads per 2 WMMAs). 8 k-steps over K=256.
// ---------------------------------------------------------------------------
__global__ void gemm_k(const unsigned short* __restrict__ h,
                       const unsigned int* __restrict__ pb,
                       float* __restrict__ W,
                       int ebase, int NW, int ntiles, int npairs) {
    int lane = threadIdx.x & 31, wv = threadIdx.x >> 5;
    int p = blockIdx.y * 4 + wv;
    if (p >= npairs) return;                        // wave-uniform: EXEC stays all-1
    int nt0 = 2*p;
    int hf = lane >> 4, ml = lane & 15;
    const unsigned short* arow = h + (size_t)(ebase + blockIdx.x*16 + ml)*256 + 16*hf;
    const unsigned int*   bb0 = pb + ((size_t)nt0*256 + (size_t)lane)*8;
    const unsigned int*   bb1 = bb0 + 2048;         // next 16-col tile
    v8f acc0 = {0.f,0.f,0.f,0.f,0.f,0.f,0.f,0.f};
    v8f acc1 = {0.f,0.f,0.f,0.f,0.f,0.f,0.f,0.f};
    #pragma unroll
    for (int ks = 0; ks < 8; ++ks) {
        union U { uint4 q[2]; v16bf v; } a, b0, b1;
        a.q[0] = *(const uint4*)(arow + ks*32);
        a.q[1] = *(const uint4*)(arow + ks*32 + 8);
        const unsigned int* p0 = bb0 + (size_t)ks*256;
        const unsigned int* p1 = bb1 + (size_t)ks*256;
        b0.q[0] = *(const uint4*)(p0);  b0.q[1] = *(const uint4*)(p0 + 4);
        b1.q[0] = *(const uint4*)(p1);  b1.q[1] = *(const uint4*)(p1 + 4);
        acc0 = __builtin_amdgcn_wmma_f32_16x16x32_bf16(false, a.v, false, b0.v,
                                                       (short)0, acc0, false, false);
        acc1 = __builtin_amdgcn_wmma_f32_16x16x32_bf16(false, a.v, false, b1.v,
                                                       (short)0, acc1, false, false);
    }
    int col0 = nt0*16 + ml;
    int r0   = blockIdx.x*16 + 8*hf;
    #pragma unroll
    for (int d = 0; d < 8; ++d)
        W[(size_t)(r0 + d)*NW + col0] = acc0[d];
    if (nt0 + 1 < ntiles) {                         // wave-uniform guard (padded tile)
        #pragma unroll
        for (int d = 0; d < 8; ++d)
            W[(size_t)(r0 + d)*NW + col0 + 16] = acc1[d];
    }
}

// ---------------------------------------------------------------------------
// Kernel 5: per-edge tensor product + scatter (sparse w3j contraction).
// ---------------------------------------------------------------------------
__global__ void tp_k(const float* __restrict__ xin, const float* __restrict__ shb,
                     const float* __restrict__ W,
                     const int* __restrict__ src, const int* __restrict__ dst,
                     float* __restrict__ xout,
                     int ebase, int cM, int NW, int din, int dout,
                     int is0, int icnt) {
    int t = blockIdx.x * blockDim.x + threadIdx.x;
    if (t >= cM) return;
    int e = ebase + t;
    float xl[128], shv[16], out[160];
    const float* xrow = xin + (size_t)src[e]*din;
    for (int i = 0; i < din; ++i) xl[i] = xrow[i];
    #pragma unroll
    for (int i = 0; i < 16; ++i) shv[i] = shb[(size_t)e*16 + i];
    for (int i = 0; i < dout; ++i) out[i] = 0.0f;
    const float* wrow = W + (size_t)t*NW;
    for (int ii = is0; ii < is0 + icnt; ++ii) {
        TPI in = TP_INSTR[ii];
        int st = W3S[in.combo], cn = W3N[in.combo];
        for (int u = 0; u < in.m1; ++u) {
            const float* s1 = xl + in.x1o + u*in.d1;
            float z[3] = {0.f, 0.f, 0.f};
            for (int k = st; k < st + cn; ++k)
                z[W3C[k]] += W3V[k] * s1[W3A[k]] * shv[in.sho + W3B[k]];
            const float* wp = wrow + in.woff + u*in.mo;
            for (int w = 0; w < in.mo; ++w) {
                float ww = wp[w] * in.alpha;
                int o = in.outo + w*in.d3;
                out[o] += ww * z[0];
                if (in.d3 > 1) { out[o+1] += ww*z[1]; out[o+2] += ww*z[2]; }
            }
        }
    }
    int d = dst[e];
    for (int i = 0; i < dout; ++i)
        atomicAdd(&xout[(size_t)d*dout + i], out[i]*INV_SQRT_NEIGH);
}

// ---------------------------------------------------------------------------
// Kernel 6: gate nonlinearity. scal [A relu, A abs]; gates [B relu, B tanh]x2;
// gated two (2B, l=1) blocks.
// ---------------------------------------------------------------------------
__global__ void gate_k(const float* __restrict__ xin, float* __restrict__ xout,
                       int A, int B, int N) {
    int n = blockIdx.x * blockDim.x + threadIdx.x;
    if (n >= N) return;
    int C = 2*B, din = 2*A + 16*B, dout = 2*A + 12*B;
    const float* x = xin + (size_t)n*din;
    float* y = xout + (size_t)n*dout;
    for (int i = 0; i < A; ++i) { y[i] = fmaxf(x[i], 0.0f); y[A+i] = fabsf(x[A+i]); }
    int gb = 2*A, vb = 2*A + 4*B;
    for (int m = 0; m < C; ++m) {
        float g1 = (m < B) ? fmaxf(x[gb + m],     0.0f) : tanhf(x[gb + m]);
        float g2 = (m < B) ? fmaxf(x[gb + C + m], 0.0f) : tanhf(x[gb + C + m]);
        for (int c = 0; c < 3; ++c) {
            y[2*A + m*3 + c]       = x[vb + m*3 + c]       * g1;
            y[2*A + 3*C + m*3 + c] = x[vb + 3*C + m*3 + c] * g2;
        }
    }
}

__global__ void readout_k(const float* __restrict__ x, const int* __restrict__ batch,
                          float* __restrict__ gacc, int N) {
    int n = blockIdx.x * blockDim.x + threadIdx.x;
    if (n >= N) return;
    atomicAdd(&gacc[batch[n]], x[n]);
}

__global__ void sigmoid_k(const float* __restrict__ gacc, float* __restrict__ out) {
    int g = threadIdx.x;
    if (g < N_GRAPHS) out[g] = 1.0f / (1.0f + __expf(-0.5f*gacc[g]));
}

// ---------------------------------------------------------------------------
// Host orchestration
// ---------------------------------------------------------------------------
extern "C" void kernel_launch(void* const* d_in, const int* in_sizes, int n_in,
                              void* d_out, int out_size, void* d_ws, size_t ws_size,
                              hipStream_t stream) {
    (void)in_sizes; (void)n_in; (void)out_size;
    const int N = N_NODES, E = N_EDGES;
    const float* pos   = (const float*)d_in[0];
    const int*   esrc  = (const int*)d_in[1];
    const int*   edst  = (const int*)d_in[2];
    const int*   batch = (const int*)d_in[3];

    // workspace layout (256B aligned, W takes the remainder)
    char* ws = (char*)d_ws;
    size_t cur = 0;
    auto alloc = [&](size_t bytes) {
        void* p = ws + cur;
        cur += (bytes + 255) & ~(size_t)255;
        return p;
    };
    float*          shb  = (float*)alloc((size_t)E*16*4);
    float*          esb  = (float*)alloc((size_t)E*3*4);
    unsigned short* hb   = (unsigned short*)alloc((size_t)E*256*2);
    unsigned int*   pbb  = (unsigned int*)alloc((size_t)3328*128*4);
    float*          xA   = (float*)alloc((size_t)N*160*4);
    float*          xB   = (float*)alloc((size_t)N*160*4);
    float*          gacc = (float*)alloc(256);
    size_t wavail = (ws_size > cur) ? (ws_size - cur) : 0;
    // Clamp the W staging chunk to 64MB so GEMM stores + TP loads stay L2-resident
    // (MI455X global L2 = 192MB); avoids the HBM round-trip for the dominant traffic.
    const size_t W_L2_CAP = (size_t)64 << 20;
    if (wavail > W_L2_CAP) wavail = W_L2_CAP;
    float* Wbuf = (float*)(ws + cur);

    hipMemsetAsync(xA, 0, (size_t)N*16*4, stream);
    hipMemsetAsync(gacc, 0, 256, stream);
    edge_prep_k<<<(E + 255)/256, 256, 0, stream>>>(pos, esrc, edst, shb, esb, xA, E);

    struct Layer { int NW, ntiles, din, dconv, is0, icnt, A, B; };
    const Layer Ls[4] = {
        { 272,  17,  16, 160,  0, 21, 16, 8},
        {3328, 208, 128, 128, 21, 20, 16, 6},
        {2400, 150, 104, 104, 41, 20, 20, 4},
        {  48,   3,  88,   1, 61,  3,  0, 0}};

    for (int l = 0; l < 4; ++l) {
        const Layer& L = Ls[l];
        const float* w1 = (const float*)d_in[4 + 2*l];
        const float* w2 = (const float*)d_in[5 + 2*l];

        h_k<<<(E*256 + 255)/256, 256, 0, stream>>>(esb, w1, hb, E);
        int ntP = (L.ntiles + 1) & ~1;              // pad to even tile count
        int npairs = ntP / 2;
        int pbTotal = ntP * 2048;                   // dwords
        packB_k<<<(pbTotal + 255)/256, 256, 0, stream>>>(w2, pbb, L.NW, pbTotal);
        hipMemsetAsync(xB, 0, (size_t)N*L.dconv*4, stream);

        int chunk = (int)(wavail / ((size_t)L.NW * 4));
        chunk &= ~15;
        if (chunk < 16) chunk = 16;
        if (chunk > E)  chunk = E;

        for (int base = 0; base < E; base += chunk) {
            int cM = (E - base < chunk) ? (E - base) : chunk;     // multiple of 16
            dim3 gg(cM/16, (npairs + 3)/4);
            gemm_k<<<gg, 128, 0, stream>>>(hb, pbb, Wbuf, base, L.NW, L.ntiles, npairs);
            tp_k<<<(cM + 255)/256, 256, 0, stream>>>(xA, shb, Wbuf, esrc, edst, xB,
                                                     base, cM, L.NW, L.din, L.dconv,
                                                     L.is0, L.icnt);
        }
        if (l < 3)
            gate_k<<<(N + 255)/256, 256, 0, stream>>>(xB, xA, L.A, L.B, N);
        else
            readout_k<<<(N + 255)/256, 256, 0, stream>>>(xB, batch, gacc, N);
    }
    sigmoid_k<<<1, 64, 0, stream>>>(gacc, (float*)d_out);
}
